// HabitatInteraction_32847909880496
// MI455X (gfx1250) — matrix operand order repaired and checked by reading, not compiled, not run on hardware
//
#include <hip/hip_runtime.h>
#include <math.h>

// ---------------- problem constants (from reference) ----------------
#define NB    64        // graphs
#define NPG   500       // nodes per graph
#define CH    128       // channels C
#define PEDIM 8
#define KCL   2         // clusters K
#define NHEAD 4
#define NLAY  3
#define NN    (NB*NPG)  // 32000 nodes
#define EPG   8000
#define NE    (NB*EPG)  // 512000 edges
#define XDIM  120
#define HD    32        // head dim
#define QT    32        // ceil(NPG/16) query tiles

typedef __attribute__((ext_vector_type(16))) __bf16 v16bf;
typedef __attribute__((ext_vector_type(8)))  __bf16 bf16x8;
typedef __attribute__((ext_vector_type(8)))  float  v8f;

// Build a v16bf fragment from two contiguous 8-element bf16 chunks.
// A-fragment (16x32): lane m=lane&15, half=lane>>4;
//   elements 0..15 = A[m][k0+half*16 .. +15]
// B-fragment (32x16): lane n=lane&15; elements 0..15 = K = k0+half*16 .. +15
__device__ __forceinline__ v16bf ld2x8(const __bf16* p0, const __bf16* p1) {
  bf16x8 a = *(const bf16x8*)p0;
  bf16x8 b = *(const bf16x8*)p1;
  v16bf r;
#pragma unroll
  for (int i = 0; i < 8; ++i) { r[i] = a[i]; r[8 + i] = b[i]; }
  return r;
}

// ================= operand packing =================
// Apk[i] = bf16(A[i] (+ A2[i]))   (fuses the residual add outside the hot loop)
__global__ void pack_a(const float* __restrict__ A, const float* __restrict__ A2,
                       __bf16* __restrict__ Apk, long long n)
{
  long long i = (long long)blockIdx.x * blockDim.x + threadIdx.x;
  if (i >= n) return;
  float v = A[i];
  if (A2) v += A2[i];
  Apk[i] = (__bf16)v;
}

// Wp[c*Kd + k] = bf16(W[k*Nc + c])  (transpose so B fragments are contiguous)
__global__ void pack_w(const float* __restrict__ W, __bf16* __restrict__ Wp,
                       int Kd, int Nc)
{
  int i = blockIdx.x * blockDim.x + threadIdx.x;
  if (i >= Kd * Nc) return;
  int k = i / Nc, c = i % Nc;
  Wp[(size_t)c * Kd + k] = (__bf16)W[i];
}

// ================= bf16 WMMA GEMM core, 16x64 per wave =================
// Out[M x Nc] = act( Apk[M x Kd] * W + bias )
// One wave computes a 16(M) x 64(N) strip: the A fragment is loaded once per
// k-step and reused by 4 WMMAs (2 A-loads + 8 B-loads per 4 wmma = 2.5 ld/wmma).
__global__ void gemm_core(const __bf16* __restrict__ Ap, const __bf16* __restrict__ Wp,
                          const float* __restrict__ bias, float* __restrict__ Out,
                          int M, int Kd, int Nc, int relu)
{
  int wave  = blockIdx.x * (blockDim.x >> 5) + (threadIdx.x >> 5);
  int tN64  = Nc >> 6;
  int tM    = M >> 4;
  if (wave >= tM * tN64) return;
  int tm = wave / tN64, tn = wave % tN64;
  int lane = threadIdx.x & 31;
  int l16  = lane & 15;
  int half = lane >> 4;
  const __bf16* arow = Ap + (size_t)(tm * 16 + l16) * Kd;
  const __bf16* wc0  = Wp + (size_t)(tn * 64 + l16) * Kd;
  const __bf16* wc1  = wc0 + (size_t)16 * Kd;
  const __bf16* wc2  = wc0 + (size_t)32 * Kd;
  const __bf16* wc3  = wc0 + (size_t)48 * Kd;
  v8f acc0 = {}, acc1 = {}, acc2 = {}, acc3 = {};
  for (int k0 = 0; k0 < Kd; k0 += 32) {
    v16bf af = ld2x8(arow + k0 + half * 16, arow + k0 + half * 16 + 8);
    int kb = k0 + half * 16;
    v16bf b0 = ld2x8(wc0 + kb, wc0 + kb + 8);
    v16bf b1 = ld2x8(wc1 + kb, wc1 + kb + 8);
    v16bf b2 = ld2x8(wc2 + kb, wc2 + kb + 8);
    v16bf b3 = ld2x8(wc3 + kb, wc3 + kb + 8);
    acc0 = __builtin_amdgcn_wmma_f32_16x16x32_bf16(false, af, false, b0, (short)0, acc0, false, false);
    acc1 = __builtin_amdgcn_wmma_f32_16x16x32_bf16(false, af, false, b1, (short)0, acc1, false, false);
    acc2 = __builtin_amdgcn_wmma_f32_16x16x32_bf16(false, af, false, b2, (short)0, acc2, false, false);
    acc3 = __builtin_amdgcn_wmma_f32_16x16x32_bf16(false, af, false, b3, (short)0, acc3, false, false);
  }
#pragma unroll
  for (int j = 0; j < 4; ++j) {
    v8f acc = (j == 0) ? acc0 : (j == 1) ? acc1 : (j == 2) ? acc2 : acc3;
    int col = tn * 64 + j * 16 + l16;
    float bv = bias ? bias[col] : 0.0f;
#pragma unroll
    for (int r = 0; r < 8; ++r) {
      int orow = tm * 16 + half * 8 + r;   // D layout: lanes 16-31 hold M=8..15
      float v = acc[r] + bv;
      if (relu) v = fmaxf(v, 0.0f);
      Out[(size_t)orow * Nc + col] = v;
    }
  }
}

// ================= attention: one wave per (graph, head, q-tile) =================
// qkvb: [NN x 384] bf16 (q:0-127, k:128-255, v:256-383); o: [NN x 128] f32
// LDS phases (48 KB): [scores f32 16x512 | probs bf16 16x512] ->
//                     [V^T  bf16 32x512 | probs bf16 16x512]
__global__ void attn_fa(const __bf16* __restrict__ qkvb, float* __restrict__ o)
{
  __shared__ char smem[32768 + 16384];
  float*  sc = (float*)smem;             // 16 x 512 f32 scores (phase 1)
  __bf16* vt = (__bf16*)smem;            // 32 x 512 bf16 V^T (phase 2, overlays sc)
  __bf16* pb = (__bf16*)(smem + 32768);  // 16 x 512 bf16 probabilities

  int wid = blockIdx.x;
  int qt = wid & 31;
  int hh = (wid >> 5) & 3;
  int b  = wid >> 7;
  int lane = threadIdx.x & 31;
  int l16 = lane & 15, half = lane >> 4;

  int qrow = qt * 16 + l16; if (qrow > NPG - 1) qrow = NPG - 1;
  const __bf16* qp = qkvb + (size_t)(b * NPG + qrow) * 384 + hh * HD;
  v16bf aq = ld2x8(qp + half * 16, qp + half * 16 + 8);
  const float scale = 0.17677669529663687f;  // 1/sqrt(32)

  for (int t = 0; t < QT; ++t) {
    int key = t * 16 + l16;
    bool kv = key < NPG;
    const __bf16* kp = qkvb + (size_t)(b * NPG + (kv ? key : 0)) * 384 + 128 + hh * HD
                     + half * 16;
    v16bf bk = ld2x8(kp, kp + 8);
    v8f s = {};
    s = __builtin_amdgcn_wmma_f32_16x16x32_bf16(false, aq, false, bk, (short)0, s, false, false);
#pragma unroll
    for (int r = 0; r < 8; ++r) {
      int row = half * 8 + r;
      sc[row * 512 + t * 16 + l16] = kv ? s[r] * scale : -1e30f;
    }
  }
  __syncthreads();
  if (lane < 16) {   // per-row softmax over 512 padded keys -> bf16 probs
    float mx = -1e30f;
    for (int j = 0; j < 512; ++j) mx = fmaxf(mx, sc[lane * 512 + j]);
    float sum = 0.f;
    for (int j = 0; j < 512; ++j) { float p = __expf(sc[lane * 512 + j] - mx); sc[lane * 512 + j] = p; sum += p; }
    float inv = 1.0f / sum;
    for (int j = 0; j < 512; ++j) pb[lane * 512 + j] = (__bf16)(sc[lane * 512 + j] * inv);
  }
  __syncthreads();   // pb done; sc storage now dead -> reuse as V^T

  // stage V^T into LDS: vt[dim*512 + key] = V[key][dim]; each lane does 8 key-pairs
  for (int pidx = lane; pidx < 256; pidx += 32) {
    int k0 = pidx * 2, k1 = k0 + 1;
    const bf16x8* p0 = (const bf16x8*)(qkvb + (size_t)(b * NPG + (k0 < NPG ? k0 : 0)) * 384 + 256 + hh * HD);
    const bf16x8* p1 = (const bf16x8*)(qkvb + (size_t)(b * NPG + (k1 < NPG ? k1 : 0)) * 384 + 256 + hh * HD);
    bf16x8 r0[4], r1[4];
#pragma unroll
    for (int j = 0; j < 4; ++j) {
      bf16x8 z = {};
      r0[j] = (k0 < NPG) ? p0[j] : z;
      r1[j] = (k1 < NPG) ? p1[j] : z;
    }
#pragma unroll
    for (int d = 0; d < 32; ++d) {
      vt[d * 512 + k0] = r0[d >> 3][d & 7];
      vt[d * 512 + k1] = r1[d >> 3][d & 7];
    }
  }
  __syncthreads();

  v8f acc0 = {}, acc1 = {};
  for (int kc = 0; kc < 16; ++kc) {      // P(16x512) * V(512x32), all from LDS
    const __bf16* pp = pb + l16 * 512 + kc * 32;
    v16bf ap = ld2x8(pp + half * 16, pp + half * 16 + 8);
    const __bf16* v0p = vt + (size_t)l16 * 512 + kc * 32 + half * 16;
    const __bf16* v1p = vt + (size_t)(16 + l16) * 512 + kc * 32 + half * 16;
    v16bf bv0 = ld2x8(v0p, v0p + 8);
    v16bf bv1 = ld2x8(v1p, v1p + 8);
    acc0 = __builtin_amdgcn_wmma_f32_16x16x32_bf16(false, ap, false, bv0, (short)0, acc0, false, false);
    acc1 = __builtin_amdgcn_wmma_f32_16x16x32_bf16(false, ap, false, bv1, (short)0, acc1, false, false);
  }
#pragma unroll
  for (int r = 0; r < 8; ++r) {
    int row = half * 8 + r;
    int q = qt * 16 + row;
    if (q < NPG) {
      float* op = o + (size_t)(b * NPG + q) * CH + hh * HD;
      op[l16]      = acc0[r];
      op[16 + l16] = acc1[r];
    }
  }
}

// ================= PE embedding path =================
__global__ void pe_gather(const int* __restrict__ pe, const float* __restrict__ emb,
                          float* __restrict__ raw)
{
  int n = blockIdx.x * blockDim.x + threadIdx.x;
  if (n >= NN) return;
  int idx = pe[n];
  for (int j = 0; j < 20; ++j) raw[n * 20 + j] = emb[idx * 20 + j];
}

__global__ void pe_fuse(const float* __restrict__ x, const float* __restrict__ raw,
                        const float* __restrict__ st, const float* __restrict__ g,
                        const float* __restrict__ bb, const float* __restrict__ w,
                        const float* __restrict__ wb, float* __restrict__ h)
{
  int n = blockIdx.x * blockDim.x + threadIdx.x;
  if (n >= NN) return;
  float xn[20];
  for (int j = 0; j < 20; ++j) {
    float mu = st[j] / NN;
    float var = st[20 + j] / NN - mu * mu;
    xn[j] = (raw[n * 20 + j] - mu) * rsqrtf(var + 1e-5f) * g[j] + bb[j];
  }
  float* hr = h + (size_t)n * CH;
  for (int j = 0; j < XDIM; ++j) hr[j] = x[(size_t)n * XDIM + j];
  for (int oo = 0; oo < PEDIM; ++oo) {
    float acc = wb[oo];
    for (int j = 0; j < 20; ++j) acc += xn[j] * w[j * PEDIM + oo];
    hr[XDIM + oo] = acc;
  }
}

// ================= GINE edge scatter (fused edge-lin + relu + segment-sum) ====
__global__ void edge_scatter(const float* __restrict__ h, const int* __restrict__ ei,
                             const float* __restrict__ ea, const float* __restrict__ ew,
                             const float* __restrict__ ebv, float* __restrict__ aggr)
{
  long long tid = (long long)blockIdx.x * blockDim.x + threadIdx.x;
  if (tid >= (long long)NE * 32) return;
  int e = (int)(tid >> 5);
  int c0 = ((int)(tid & 31)) * 4;
  int src = ei[e], dst = ei[NE + e];
  float a = ea[e];
  const float* hs = h + (size_t)src * CH + c0;
  float* ag = aggr + (size_t)dst * CH + c0;
  __builtin_prefetch(hs, 0, 1);   // global_prefetch_b8
#pragma unroll
  for (int j = 0; j < 4; ++j) {
    float m = fmaxf(hs[j] + a * ew[c0 + j] + ebv[c0 + j], 0.0f);
    atomicAdd(&ag[j], m);
  }
}

// ================= batchnorm column stats / apply =================
__global__ void col_stats(const float* __restrict__ A, const float* __restrict__ A2,
                          float* __restrict__ st, int rows, int cols)
{
  int col = blockIdx.x;
  float s = 0.f, ss = 0.f;
  for (int r = blockIdx.y * blockDim.x + threadIdx.x; r < rows; r += blockDim.x * gridDim.y) {
    float v = A[(size_t)r * cols + col];
    if (A2) v += A2[(size_t)r * cols + col];
    s += v; ss += v * v;
  }
  __shared__ float sh[512];
  sh[threadIdx.x] = s; sh[256 + threadIdx.x] = ss;
  __syncthreads();
  for (int off = blockDim.x >> 1; off > 0; off >>= 1) {
    if (threadIdx.x < (unsigned)off) {
      sh[threadIdx.x] += sh[threadIdx.x + off];
      sh[256 + threadIdx.x] += sh[256 + threadIdx.x + off];
    }
    __syncthreads();
  }
  if (threadIdx.x == 0) { atomicAdd(&st[col], sh[0]); atomicAdd(&st[cols + col], sh[256]); }
}

__global__ void bn_apply(const float* __restrict__ A, const float* __restrict__ A2,
                         const float* __restrict__ st, const float* __restrict__ g,
                         const float* __restrict__ bb, float* __restrict__ Out,
                         int rows, int cols)
{
  size_t i = (size_t)blockIdx.x * blockDim.x + threadIdx.x;
  if (i >= (size_t)rows * cols) return;
  int c = (int)(i % cols);
  float mu = st[c] / rows;
  float var = st[cols + c] / rows - mu * mu;
  float v = A[i]; if (A2) v += A2[i];
  Out[i] = (v - mu) * rsqrtf(var + 1e-5f) * g[c] + bb[c];
}

__global__ void addk(const float* __restrict__ A, const float* __restrict__ Bp,
                     float* __restrict__ O, long long n)
{
  long long i = (long long)blockIdx.x * blockDim.x + threadIdx.x;
  if (i < n) O[i] = A[i] + Bp[i];
}

// ================= aggregation / losses =================
__global__ void apag(const float* __restrict__ h, const float* __restrict__ as,
                     float* __restrict__ ap, float* __restrict__ ag)
{
  int b = blockIdx.x, c = threadIdx.x;   // blockDim = 128
  float sa = 0.f, sg = 0.f;
  for (int n = 0; n < NPG; ++n) {
    int node = b * NPG + n;
    float v = h[(size_t)node * CH + c];
    sa += as[node * 2] * v;
    sg += v;
  }
  ap[b * CH + c] = sa;
  ag[b * CH + c] = sg * (1.0f / NPG);
}

__global__ void losses(const float* __restrict__ as, float* __restrict__ sbuf)
{
  int n = blockIdx.x * blockDim.x + threadIdx.x;
  float ent = 0.f, ab = 0.f, a0 = 0.f;
  if (n < NN) {
    float x0 = as[n * 2], x1 = as[n * 2 + 1];
    float p0 = fmaxf(x0, 1e-8f), p1 = fmaxf(x1, 1e-8f);
    ent = -(p0 * logf(p0) + p1 * logf(p1));
    ab = fabsf(x0) + fabsf(x1);
    a0 = x0;
  }
  __shared__ float sh[3 * 256];
  int t = threadIdx.x;
  sh[t] = ent; sh[256 + t] = ab; sh[512 + t] = a0;
  __syncthreads();
  for (int off = 128; off > 0; off >>= 1) {
    if (t < off) { sh[t] += sh[t + off]; sh[256 + t] += sh[256 + t + off]; sh[512 + t] += sh[512 + t + off]; }
    __syncthreads();
  }
  if (t == 0) { atomicAdd(&sbuf[0], sh[0]); atomicAdd(&sbuf[1], sh[256]); atomicAdd(&sbuf[2], sh[512]); }
}

// new_adj[b,k,l] = sum_edges a[src,k]*a[dst,l]  (K=2 -> 4 scalars per graph)
__global__ void newadj_k(const int* __restrict__ ei, const float* __restrict__ as,
                         float* __restrict__ nadj)
{
  int e = blockIdx.x * blockDim.x + threadIdx.x;
  if (e >= NE) return;
  int s = ei[e], d = ei[NE + e];
  int b = s / NPG;
  float s0 = as[s * 2], s1 = as[s * 2 + 1];
  float d0 = as[d * 2], d1 = as[d * 2 + 1];
  atomicAdd(&nadj[b * 4 + 0], s0 * d0);
  atomicAdd(&nadj[b * 4 + 1], s0 * d1);
  atomicAdd(&nadj[b * 4 + 2], s1 * d0);
  atomicAdd(&nadj[b * 4 + 3], s1 * d1);
}

__global__ void finalize_k(const float* __restrict__ sbuf, const float* __restrict__ nadj,
                           float* __restrict__ out_pos, float* __restrict__ out_ent)
{
  __shared__ float sh[256];
  int t = threadIdx.x;
  float p = 0.f;
  for (int i = t; i < NB * 2; i += blockDim.x) {
    int b = i >> 1, k = i & 1;
    float denom = fabsf(nadj[b * 4 + k]) + fabsf(nadj[b * 4 + 2 + k]);
    float diag = nadj[b * 4 + k * 2 + k] / fmaxf(denom, 1e-5f);
    p += (diag - 1.f) * (diag - 1.f);
  }
  sh[t] = p;
  __syncthreads();
  for (int off = 128; off > 0; off >>= 1) { if (t < off) sh[t] += sh[t + off]; __syncthreads(); }
  if (t == 0) {
    float pos = sh[0] / (NB * 2);
    float m0 = sbuf[2] / NN - 0.2f;
    float ent = sbuf[0] / NN + sbuf[1] / ((float)NN * KCL) + m0 * m0 + pos;
    *out_pos = pos;
    *out_ent = ent;
  }
}

__global__ void final_lin2(const float* __restrict__ hrelu, const float* __restrict__ w,
                           const float* __restrict__ bb, float* __restrict__ logits)
{
  int i = blockIdx.x * blockDim.x + threadIdx.x;
  if (i >= NB * KCL) return;
  int b = i >> 1, k = i & 1;
  float acc = bb[k];
  for (int j = 0; j < CH; ++j) acc += hrelu[b * CH + j] * w[j * KCL + k];
  logits[i] = acc;
}

// ================= driver =================
extern "C" void kernel_launch(void* const* d_in, const int* in_sizes, int n_in,
                              void* d_out, int out_size, void* d_ws, size_t ws_size,
                              hipStream_t stream)
{
  (void)in_sizes; (void)n_in; (void)out_size; (void)ws_size;

  // inputs in setup_inputs() dict (insertion) order, params flattened depth-first
  const float* x      = (const float*)d_in[0];
  const int*   pe     = (const int*)d_in[1];
  const int*   ei     = (const int*)d_in[2];
  const float* ea     = (const float*)d_in[3];
  const float* as     = (const float*)d_in[5];
  const float* pe_emb = (const float*)d_in[6];
  const float* pe_g   = (const float*)d_in[7];
  const float* pe_b   = (const float*)d_in[8];
  const float* pl_w   = (const float*)d_in[9];
  const float* pl_b   = (const float*)d_in[10];
  const float* l1w    = (const float*)d_in[11];
  const float* l1b    = (const float*)d_in[12];
  const float* l2w    = (const float*)d_in[13];
  const float* l2b    = (const float*)d_in[14];

  float* ws = (float*)d_ws;
  float* h  = ws;                        // N x 128 persistent node features
  float* tA = h  + (size_t)NN * CH;      // N x 384 (qkv / mlp hidden)
  float* tB = tA + (size_t)NN * 384;     // N x 128
  float* tC = tB + (size_t)NN * CH;      // N x 128
  float* tD = tC + (size_t)NN * CH;      // N x 128
  float* st = tD + (size_t)NN * CH;      // BN column sums/sumsq
  float* sbuf = st + 1024;               // [0..2]=loss accumulators
  float* nadj = sbuf + 16;               // B x 4 coarsened adjacency
  __bf16* Apk = (__bf16*)(sbuf + 4096);  // N x 384 bf16 packed A / qkv
  __bf16* Wpk = Apk + (size_t)NN * 384;  // up to 128x384 bf16 packed weights

  float* out      = (float*)d_out;
  float* o_logits = out;                 // 128
  float* o_ap     = out + 128;           // 8192
  float* o_ag     = out + 128 + 8192;    // 8192
  float* o_pos    = out + 16512;
  float* o_ent    = out + 16513;
  float* o_as     = out + 16514;         // 64000

  auto gemm = [&](const float* A, const float* A2, const float* W, const float* bias,
                  float* O, int M, int Kd, int Nc, int relu) {
    long long na = (long long)M * Kd;
    pack_a<<<(int)((na + 255) / 256), 256, 0, stream>>>(A, A2, Apk, na);
    pack_w<<<(Kd * Nc + 255) / 256, 256, 0, stream>>>(W, Wpk, Kd, Nc);
    int waves = (M / 16) * (Nc / 64);
    gemm_core<<<(waves + 7) / 8, 256, 0, stream>>>(Apk, Wpk, bias, O, M, Kd, Nc, relu);
  };

  // ---- PE embedding -> batchnorm -> linear -> concat into h ----
  hipMemsetAsync(st, 0, 1024 * sizeof(float), stream);
  pe_gather<<<(NN + 255) / 256, 256, 0, stream>>>(pe, pe_emb, tC);
  col_stats<<<dim3(20, 16), 256, 0, stream>>>(tC, nullptr, st, NN, 20);
  pe_fuse<<<(NN + 255) / 256, 256, 0, stream>>>(x, tC, st, pe_g, pe_b, pl_w, pl_b, h);

  // ---- 3 GPS layers ----
  for (int L = 0; L < NLAY; ++L) {
    int base = 15 + L * 20;
    const float* g1w = (const float*)d_in[base + 0];
    const float* g1b = (const float*)d_in[base + 1];
    const float* g2w = (const float*)d_in[base + 2];
    const float* g2b = (const float*)d_in[base + 3];
    const float* elw = (const float*)d_in[base + 4];
    const float* elb = (const float*)d_in[base + 5];
    const float* aiw = (const float*)d_in[base + 6];
    const float* aib = (const float*)d_in[base + 7];
    const float* aow = (const float*)d_in[base + 8];
    const float* aob = (const float*)d_in[base + 9];
    const float* m1w = (const float*)d_in[base + 10];
    const float* m1b = (const float*)d_in[base + 11];
    const float* m2w = (const float*)d_in[base + 12];
    const float* m2b = (const float*)d_in[base + 13];
    const float* b1g = (const float*)d_in[base + 14];
    const float* b1b = (const float*)d_in[base + 15];
    const float* b2g = (const float*)d_in[base + 16];
    const float* b2b = (const float*)d_in[base + 17];
    const float* b3g = (const float*)d_in[base + 18];
    const float* b3b = (const float*)d_in[base + 19];

    // GINE: aggr -> tB; lin2(relu(lin1(h+aggr))) -> tD
    hipMemsetAsync(tB, 0, (size_t)NN * CH * sizeof(float), stream);
    edge_scatter<<<(NE * 32) / 256, 256, 0, stream>>>(h, ei, ea, elw, elb, tB);
    gemm(h, tB, g1w, g1b, tC, NN, CH, CH, 1);
    gemm(tC, nullptr, g2w, g2b, tD, NN, CH, CH, 0);

    // MHA: qkv -> tA (f32); pack qkv -> Apk (bf16); attention -> tB; out-proj -> tC
    gemm(h, nullptr, aiw, aib, tA, NN, CH, 384, 0);
    pack_a<<<(int)(((long long)NN * 384 + 255) / 256), 256, 0, stream>>>(tA, nullptr, Apk,
                                                                         (long long)NN * 384);
    attn_fa<<<NB * NHEAD * QT, 32, 0, stream>>>(Apk, tB);
    gemm(tB, nullptr, aow, aob, tC, NN, CH, CH, 0);

    // h1 = BN1(gine + h) -> tD ; h2 = BN2(attn + h) -> tC
    hipMemsetAsync(st, 0, 1024 * sizeof(float), stream);
    col_stats<<<dim3(CH, 16), 256, 0, stream>>>(tD, h, st, NN, CH);
    bn_apply<<<(NN * CH + 255) / 256, 256, 0, stream>>>(tD, h, st, b1g, b1b, tD, NN, CH);
    hipMemsetAsync(st, 0, 1024 * sizeof(float), stream);
    col_stats<<<dim3(CH, 16), 256, 0, stream>>>(tC, h, st, NN, CH);
    bn_apply<<<(NN * CH + 255) / 256, 256, 0, stream>>>(tC, h, st, b2g, b2b, tC, NN, CH);

    // out = h1 + h2 -> tB ; MLP -> tC ; h = BN3(out + mlp)
    addk<<<(NN * CH + 255) / 256, 256, 0, stream>>>(tD, tC, tB, (long long)NN * CH);
    gemm(tB, nullptr, m1w, m1b, tA, NN, CH, 256, 1);
    gemm(tA, nullptr, m2w, m2b, tC, NN, 256, CH, 0);
    hipMemsetAsync(st, 0, 1024 * sizeof(float), stream);
    col_stats<<<dim3(CH, 16), 256, 0, stream>>>(tB, tC, st, NN, CH);
    bn_apply<<<(NN * CH + 255) / 256, 256, 0, stream>>>(tB, tC, st, b3g, b3b, h, NN, CH);
  }

  // ---- aggregation + losses ----
  apag<<<NB, CH, 0, stream>>>(h, as, o_ap, o_ag);
  hipMemsetAsync(sbuf, 0, 512 * sizeof(float), stream);
  losses<<<(NN + 255) / 256, 256, 0, stream>>>(as, sbuf);
  newadj_k<<<(NE + 255) / 256, 256, 0, stream>>>(ei, as, nadj);
  finalize_k<<<1, 256, 0, stream>>>(sbuf, nadj, o_pos, o_ent);
  hipMemcpyAsync(o_as, as, (size_t)NN * KCL * sizeof(float), hipMemcpyDeviceToDevice, stream);

  // ---- head: logits = lin2(relu(lin1(ap))) ----
  gemm(o_ap, nullptr, l1w, l1b, tC, NB, CH, CH, 1);
  final_lin2<<<1, 128, 0, stream>>>(tC, l2w, l2b, o_logits);
}
